// RNN_58463094833880
// MI455X (gfx1250) — matrix operand order repaired and checked by reading, not compiled, not run on hardware
//
#include <hip/hip_runtime.h>

// ---------------------------------------------------------------------------
// Tree-RNN on MI455X (gfx1250): bf16 WMMA GEMM pipeline.
//   1) fp32 -> bf16 conversion of the (small) weight matrices only.
//   2) Embedding GEMM [8192 x 4096] fp32-A (converted in-register to bf16
//      via packed v_cvt) @ W_emb^T -> relu -> h0 (bf16). Leaf data crosses
//      HBM exactly once.
//   3) 13 combine GEMMs: pair matrix == h reinterpreted [n/2, 1024] row-major
//      (siblings are adjacent rows), @ W_comb^T -> relu, ping-pong buffers.
//   4) Tiny 2x512 projection in fp32.
// All matrix math via v_wmma_f32_16x16x32_bf16 (wave32, 16x16x32).
// Wave tile 32x64 = 2x4 WMMA accumulators.
// ---------------------------------------------------------------------------

typedef __attribute__((ext_vector_type(16))) __bf16        v16bf;
typedef __attribute__((ext_vector_type(4)))  __bf16        bf16x4;
typedef __attribute__((ext_vector_type(2)))  __bf16        bf16x2;
typedef __attribute__((ext_vector_type(8)))  float         v8f;
typedef __attribute__((ext_vector_type(4)))  float         f32x4;
typedef __attribute__((ext_vector_type(2)))  float         f32x2;
typedef __attribute__((ext_vector_type(4)))  unsigned int  u32x4;

union FragBF {
    v16bf  v;
    bf16x4 h[4];
    u32x4  q[2];
};

__device__ __forceinline__ float bf2f(unsigned short h) {
    unsigned u = ((unsigned)h) << 16;
    return __builtin_bit_cast(float, u);
}

// packed pair conversion: 2 floats -> 1 dword of two bf16 (v_cvt_pk form)
__device__ __forceinline__ unsigned int pk2bf(float x, float y) {
    f32x2 f; f.x = x; f.y = y;
    bf16x2 h = __builtin_convertvector(f, bf16x2);
    return __builtin_bit_cast(unsigned int, h);
}

// ---------------------------------------------------------------------------
// fp32 -> bf16 conversion, 8 elements per thread per step (b128 in/out).
// n multiple of 8 (true for both weight matrices).
// ---------------------------------------------------------------------------
__global__ void cvt_f32_bf16_kernel(const float* __restrict__ src,
                                    unsigned int* __restrict__ dst,
                                    long n8) {
    long i = (long)blockIdx.x * blockDim.x + threadIdx.x;
    long stride = (long)gridDim.x * blockDim.x;
    for (; i < n8; i += stride) {
        f32x4 f0 = *(const f32x4*)(src + i * 8);
        f32x4 f1 = *(const f32x4*)(src + i * 8 + 4);
        u32x4 o;
        o.x = pk2bf(f0.x, f0.y);
        o.y = pk2bf(f0.z, f0.w);
        o.z = pk2bf(f1.x, f1.y);
        o.w = pk2bf(f1.z, f1.w);
        *(u32x4*)(dst + i * 4) = o;
    }
}

// ---------------------------------------------------------------------------
// Shared GEMM epilogue: C layout (32-bit C/D 16x16): VGPR r,
// lanes 0-15 -> M=r, N=lane; lanes 16-31 -> M=r+8, N=lane-16.
// ---------------------------------------------------------------------------
__device__ __forceinline__ void store_tile(const v8f acc[2][4],
                                           const float* __restrict__ bias,
                                           unsigned short* __restrict__ C,
                                           int M, int N, int tm, int tn,
                                           int la, int hi) {
    const int row_base = tm + hi * 8;
#pragma unroll
    for (int j = 0; j < 4; ++j) {
        const int col = tn + j * 16 + la;
        const float bj = bias[col];
#pragma unroll
        for (int i = 0; i < 2; ++i) {
#pragma unroll
            for (int r = 0; r < 8; ++r) {
                int row = row_base + i * 16 + r;
                if (row < M) {
                    float v = acc[i][j][r] + bj;
                    v = v > 0.f ? v : 0.f;
                    __bf16 hv = (__bf16)v;
                    C[(size_t)row * N + col] =
                        __builtin_bit_cast(unsigned short, hv);
                }
            }
        }
    }
}

// ---------------------------------------------------------------------------
// GEMM (bf16 A): C = relu(A @ B^T + bias), fp32 accumulate via WMMA.
//   A: [M x K] row-major bf16;  B: [N x K] row-major bf16;  C: [M x N] bf16
// Block: 256 threads = 8 waves; block tile 64(M) x 256(N);
// wave tile 32x64 = 2x4 WMMA accumulators.
// ---------------------------------------------------------------------------
__global__ void __launch_bounds__(256)
gemm_bf16_relu_kernel(const unsigned short* __restrict__ A,
                      const unsigned short* __restrict__ B,
                      const float* __restrict__ bias,
                      unsigned short* __restrict__ C,
                      int M, int N, int K) {
    const int bn   = blockIdx.x * 256;
    const int bm   = blockIdx.y * 64;
    const int wave = threadIdx.x >> 5;
    const int lane = threadIdx.x & 31;
    const int tm   = bm + (wave >> 2) * 32;
    const int tn   = bn + (wave & 3) * 64;
    const int la   = lane & 15;
    const int hi   = lane >> 4;

    // A frag (16-bit A 16x32): lane<16: row la, K {0..7,16..23}+k0;
    //                          lane>=16: row la, K {8..15,24..31}+k0.
    int r0 = tm + la;      if (r0 > M - 1) r0 = M - 1;
    int r1 = tm + 16 + la; if (r1 > M - 1) r1 = M - 1;
    const unsigned short* pa0 = A + (size_t)r0 * K + (hi * 8);
    const unsigned short* pa1 = A + (size_t)r1 * K + (hi * 8);

    // B frag (16-bit B 32x16): lane<16: col la, K {0..7,16..23}+k0;
    //                          lane>=16: col la, K {8..15,24..31}+k0.
    const unsigned short* pb[4];
#pragma unroll
    for (int j = 0; j < 4; ++j)
        pb[j] = B + (size_t)(tn + j * 16 + la) * K + (hi * 8);

    v8f acc[2][4] = {};

    for (int k0 = 0; k0 < K; k0 += 32) {
        FragBF a[2], b[4];
        a[0].q[0] = *(const u32x4*)(pa0 + k0);
        a[0].q[1] = *(const u32x4*)(pa0 + k0 + 16);
        a[1].q[0] = *(const u32x4*)(pa1 + k0);
        a[1].q[1] = *(const u32x4*)(pa1 + k0 + 16);
#pragma unroll
        for (int j = 0; j < 4; ++j) {
            b[j].q[0] = *(const u32x4*)(pb[j] + k0);
            b[j].q[1] = *(const u32x4*)(pb[j] + k0 + 16);
        }

        __builtin_prefetch(pa0 + k0 + 256, 0, 0);
        __builtin_prefetch(pa1 + k0 + 256, 0, 0);

#pragma unroll
        for (int i = 0; i < 2; ++i)
#pragma unroll
            for (int j = 0; j < 4; ++j)
                acc[i][j] = __builtin_amdgcn_wmma_f32_16x16x32_bf16(
                    false, a[i].v, false, b[j].v, (short)0, acc[i][j],
                    false, false);
    }

    store_tile(acc, bias, C, M, N, tm, tn, la, hi);
}

// ---------------------------------------------------------------------------
// GEMM (fp32 A, fused conversion): C = relu(A @ B^T + bias).
//   A: [M x K] row-major fp32 (leaf data; converted in-register to bf16 via
//      __builtin_convertvector -> packed v_cvt, so it crosses HBM once);
//   B: [N x K] bf16;  C: [M x N] bf16.  Same tiling as above.
// ---------------------------------------------------------------------------
__global__ void __launch_bounds__(256)
gemm_a32_bf16_relu_kernel(const float* __restrict__ A,
                          const unsigned short* __restrict__ B,
                          const float* __restrict__ bias,
                          unsigned short* __restrict__ C,
                          int M, int N, int K) {
    const int bn   = blockIdx.x * 256;
    const int bm   = blockIdx.y * 64;
    const int wave = threadIdx.x >> 5;
    const int lane = threadIdx.x & 31;
    const int tm   = bm + (wave >> 2) * 32;
    const int tn   = bn + (wave & 3) * 64;
    const int la   = lane & 15;
    const int hi   = lane >> 4;

    int r0 = tm + la;      if (r0 > M - 1) r0 = M - 1;
    int r1 = tm + 16 + la; if (r1 > M - 1) r1 = M - 1;
    const float* pa0 = A + (size_t)r0 * K + (hi * 8);
    const float* pa1 = A + (size_t)r1 * K + (hi * 8);

    const unsigned short* pb[4];
#pragma unroll
    for (int j = 0; j < 4; ++j)
        pb[j] = B + (size_t)(tn + j * 16 + la) * K + (hi * 8);

    v8f acc[2][4] = {};

    for (int k0 = 0; k0 < K; k0 += 32) {
        // fp32 A fragments: 16 floats per frag = 4 x b128
        f32x4 f00 = *(const f32x4*)(pa0 + k0);
        f32x4 f01 = *(const f32x4*)(pa0 + k0 + 4);
        f32x4 f02 = *(const f32x4*)(pa0 + k0 + 16);
        f32x4 f03 = *(const f32x4*)(pa0 + k0 + 20);
        f32x4 f10 = *(const f32x4*)(pa1 + k0);
        f32x4 f11 = *(const f32x4*)(pa1 + k0 + 4);
        f32x4 f12 = *(const f32x4*)(pa1 + k0 + 16);
        f32x4 f13 = *(const f32x4*)(pa1 + k0 + 20);

        FragBF a[2], b[4];
        a[0].h[0] = __builtin_convertvector(f00, bf16x4);
        a[0].h[1] = __builtin_convertvector(f01, bf16x4);
        a[0].h[2] = __builtin_convertvector(f02, bf16x4);
        a[0].h[3] = __builtin_convertvector(f03, bf16x4);
        a[1].h[0] = __builtin_convertvector(f10, bf16x4);
        a[1].h[1] = __builtin_convertvector(f11, bf16x4);
        a[1].h[2] = __builtin_convertvector(f12, bf16x4);
        a[1].h[3] = __builtin_convertvector(f13, bf16x4);

#pragma unroll
        for (int j = 0; j < 4; ++j) {
            b[j].q[0] = *(const u32x4*)(pb[j] + k0);
            b[j].q[1] = *(const u32x4*)(pb[j] + k0 + 16);
        }

        __builtin_prefetch(pa0 + k0 + 256, 0, 0);
        __builtin_prefetch(pa1 + k0 + 256, 0, 0);

#pragma unroll
        for (int i = 0; i < 2; ++i)
#pragma unroll
            for (int j = 0; j < 4; ++j)
                acc[i][j] = __builtin_amdgcn_wmma_f32_16x16x32_bf16(
                    false, a[i].v, false, b[j].v, (short)0, acc[i][j],
                    false, false);
    }

    store_tile(acc, bias, C, M, N, tm, tn, la, hi);
}

// ---------------------------------------------------------------------------
// Root projection: out[n] = sum_k h[k] * W_proj[n,k] + b_proj[n], n = 0,1.
// ---------------------------------------------------------------------------
__global__ void proj_kernel(const unsigned short* __restrict__ h,
                            const float* __restrict__ W,
                            const float* __restrict__ b,
                            float* __restrict__ out) {
    const int w = threadIdx.x >> 5;
    const int lane = threadIdx.x & 31;
    float s = 0.f;
    for (int k = lane; k < 512; k += 32)
        s += bf2f(h[k]) * W[w * 512 + k];
#pragma unroll
    for (int off = 16; off > 0; off >>= 1)
        s += __shfl_down(s, off, 32);
    if (lane == 0) out[w] = s + b[w];
}

// ---------------------------------------------------------------------------
extern "C" void kernel_launch(void* const* d_in, const int* in_sizes, int n_in,
                              void* d_out, int out_size, void* d_ws, size_t ws_size,
                              hipStream_t stream) {
    (void)in_sizes; (void)n_in; (void)out_size; (void)ws_size;

    const float* leaf   = (const float*)d_in[0];  // [8192, 4096]
    const float* W_emb  = (const float*)d_in[1];  // [512, 4096]
    const float* b_emb  = (const float*)d_in[2];  // [512]
    const float* W_comb = (const float*)d_in[3];  // [512, 1024]
    const float* b_comb = (const float*)d_in[4];  // [512]
    const float* W_proj = (const float*)d_in[5];  // [2, 512]
    const float* b_proj = (const float*)d_in[6];  // [2]
    float* out = (float*)d_out;                   // [2]

    const long N_LEAVES = 8192, SEQ = 4096, E = 512;

    // Workspace layout (bf16 elements), ~17 MB total:
    unsigned short* wemb_bf  = (unsigned short*)d_ws;              // 512*4096
    unsigned short* wcomb_bf = wemb_bf  + E * SEQ;                 // 512*1024
    unsigned short* hA       = wcomb_bf + E * 2 * E;               // 8192*512
    unsigned short* hB       = hA + N_LEAVES * E;                  // 4096*512

    // 1) fp32 -> bf16 staging of the weights only
    cvt_f32_bf16_kernel<<<256, 256, 0, stream>>>(
        W_emb,  (unsigned int*)wemb_bf,  E * SEQ / 8);
    cvt_f32_bf16_kernel<<<128, 256, 0, stream>>>(
        W_comb, (unsigned int*)wcomb_bf, E * 2 * E / 8);

    // 2) Embedding GEMM with fused fp32->bf16 A conversion:
    //    h0 = relu(leaf @ W_emb^T + b_emb)   [8192 x 512]
    gemm_a32_bf16_relu_kernel<<<dim3(512 / 256, 8192 / 64), 256, 0, stream>>>(
        leaf, wemb_bf, b_emb, hA, 8192, 512, 4096);

    // 3) 13 combine levels. Pair matrix [n/2, 1024] is just h viewed row-major
    //    (siblings 2i, 2i+1 are adjacent rows of h).
    unsigned short* bin  = hA;
    unsigned short* bout = hB;
    for (int n = 8192; n > 1; n >>= 1) {
        const int Mh = n >> 1;
        gemm_bf16_relu_kernel<<<dim3(512 / 256, (Mh + 63) / 64), 256, 0, stream>>>(
            bin, wcomb_bf, b_comb, bout, Mh, 512, 1024);
        unsigned short* t = bin; bin = bout; bout = t;
    }

    // 4) Root projection (fp32)
    proj_kernel<<<1, 64, 0, stream>>>(bin, W_proj, b_proj, out);
}
